// TreeModule_35519379538226
// MI455X (gfx1250) — compile-verified, matrix operand order checked by reading, not run on hardware
//
#include <hip/hip_runtime.h>

typedef __attribute__((ext_vector_type(16))) __bf16 v16bf;
typedef __attribute__((ext_vector_type(8)))  __bf16 v8bf;
typedef __attribute__((ext_vector_type(8)))  float  v8f;

#define LLEAF 8192
#define NNODE 16383
#define HDIM  256
#define EDIM  256
#define TT    32
#define G4    1024     // 4*H gate width
#define K2    512      // 2*H concat width
#define NPAD  16384

__device__ __forceinline__ float sigm(float x) { return 1.0f / (1.0f + expf(-x)); }

// ---------------- fp32 -> bf16 convert ----------------
__global__ void cvt_f32_bf16(const float* __restrict__ src, __bf16* __restrict__ dst, int n) {
  int i = blockIdx.x * blockDim.x + threadIdx.x;
  if (i < n) dst[i] = (__bf16)src[i];
}

// ---------------- embedding gather -> bf16 X[L,E] ----------------
__global__ void embed_gather(const int* __restrict__ ids, const float* __restrict__ emb,
                             __bf16* __restrict__ X) {
  int i = blockIdx.x * blockDim.x + threadIdx.x;   // LLEAF*EDIM threads
  int t = i >> 8, e = i & 255;
  X[i] = (__bf16)emb[(size_t)ids[t] * EDIM + e];
}

// ---------------- bf16 WMMA GEMM: C = A[M,K] * B[N,K]^T + bias ----------------
// Register-blocked: each wave owns a (MBLK*16) x (NBLK*16) output tile
// (MBLK*NBLK v8f accumulators). 4 waves per block stack along M.
template <int MBLK, int NBLK>
__global__ void wmma_gemm_bf16(const __bf16* __restrict__ A, int lda,
                               const __bf16* __restrict__ B, int ldb,
                               const float* __restrict__ bias,
                               const float* __restrict__ bias2,
                               float* __restrict__ C, int ldc,
                               __bf16* __restrict__ Cbf, int ldcbf,
                               int M, int N, int K) {
  const int lane  = threadIdx.x & 31;
  const int wave  = threadIdx.x >> 5;
  const int tileN = blockIdx.x * (16 * NBLK);
  const int tileM = (blockIdx.y * 4 + wave) * (16 * MBLK);
  const int m16   = lane & 15;
  const int kh    = (lane >> 4) * 8;   // K-half select per lane group

  const __bf16* arow[MBLK];
#pragma unroll
  for (int mb = 0; mb < MBLK; ++mb)
    arow[mb] = A + (size_t)(tileM + mb * 16 + m16) * lda + kh;
  const __bf16* brow[NBLK];
#pragma unroll
  for (int nb = 0; nb < NBLK; ++nb)
    brow[nb] = B + (size_t)(tileN + nb * 16 + m16) * ldb + kh;

  v8f acc[MBLK][NBLK];
#pragma unroll
  for (int mb = 0; mb < MBLK; ++mb)
#pragma unroll
    for (int nb = 0; nb < NBLK; ++nb) acc[mb][nb] = (v8f){};

  for (int k0 = 0; k0 < K; k0 += 32) {
    v16bf af[MBLK], bfv[NBLK];
#pragma unroll
    for (int mb = 0; mb < MBLK; ++mb) {
      __builtin_prefetch(arow[mb] + k0 + 128, 0, 1);
      v8bf lo = *(const v8bf*)(arow[mb] + k0);
      v8bf hi = *(const v8bf*)(arow[mb] + k0 + 16);
      af[mb] = __builtin_shufflevector(lo, hi, 0,1,2,3,4,5,6,7,8,9,10,11,12,13,14,15);
    }
#pragma unroll
    for (int nb = 0; nb < NBLK; ++nb) {
      __builtin_prefetch(brow[nb] + k0 + 128, 0, 1);
      v8bf lo = *(const v8bf*)(brow[nb] + k0);
      v8bf hi = *(const v8bf*)(brow[nb] + k0 + 16);
      bfv[nb] = __builtin_shufflevector(lo, hi, 0,1,2,3,4,5,6,7,8,9,10,11,12,13,14,15);
    }
#pragma unroll
    for (int mb = 0; mb < MBLK; ++mb)
#pragma unroll
      for (int nb = 0; nb < NBLK; ++nb)
        acc[mb][nb] = __builtin_amdgcn_wmma_f32_16x16x32_bf16(
            false, af[mb], false, bfv[nb], (short)0, acc[mb][nb], false, false);
  }

#pragma unroll
  for (int mb = 0; mb < MBLK; ++mb) {
    const int rbase = tileM + mb * 16 + (lane >> 4) * 8;
#pragma unroll
    for (int nb = 0; nb < NBLK; ++nb) {
      const int col  = tileN + nb * 16 + m16;
      const float bs = (bias ? bias[col] : 0.0f) + (bias2 ? bias2[col] : 0.0f);
#pragma unroll
      for (int r = 0; r < 8; ++r) {
        int row = rbase + r;
        if (row < M) {
          float v = acc[mb][nb][r] + bs;
          if (C)   C[(size_t)row * ldc + col] = v;
          if (Cbf) Cbf[(size_t)row * ldcbf + col] = (__bf16)v;
        }
      }
    }
  }
}

// ---------------- LSTM recurrence (persistent single workgroup) ----------------
// G[t, 1024] = x_t @ W_ih^T + b_ih + b_hh precomputed by WMMA GEMM.
// Each of 1024 threads owns one gate row of W_hh; h,c live in LDS (fp32).
__global__ void __launch_bounds__(1024)
lstm_recurrence(const float* __restrict__ G, const float* __restrict__ W_hh,
                __bf16* __restrict__ h_bf) {
  __shared__ float h_s[HDIM];
  __shared__ float c_s[HDIM];
  __shared__ float g_s[G4];
  const int tid = threadIdx.x;
  if (tid < HDIM) { h_s[tid] = 0.0f; c_s[tid] = 0.0f; }
  __syncthreads();

  const float4* wr = (const float4*)(W_hh + (size_t)tid * HDIM);
  for (int t = 0; t < LLEAF; ++t) {
    float acc = G[(size_t)t * G4 + tid];
    const float4* hv = (const float4*)h_s;
#pragma unroll 8
    for (int k = 0; k < HDIM / 4; ++k) {
      float4 w = wr[k];
      float4 h4 = hv[k];
      acc += w.x * h4.x + w.y * h4.y + w.z * h4.z + w.w * h4.w;
    }
    g_s[tid] = acc;
    __syncthreads();
    if (tid < HDIM) {
      float ig = g_s[tid], fg = g_s[HDIM + tid];
      float gg = g_s[2 * HDIM + tid], og = g_s[3 * HDIM + tid];
      float cc = sigm(fg) * c_s[tid] + sigm(ig) * tanhf(gg);
      c_s[tid] = cc;
      float hh = sigm(og) * tanhf(cc);
      h_s[tid] = hh;
      h_bf[(size_t)(LLEAF - 1 + t) * HDIM + tid] = (__bf16)hh;  // leaf node = Lc-1+t
    }
    __syncthreads();
  }
}

// ---------------- gather children h -> cat[r, 0:256]=h[2p+1], [256:512]=h[2p+2] ----------------
__global__ void gather_children(const __bf16* __restrict__ hbf, __bf16* __restrict__ cat,
                                int first, int count, int padcount) {
  int i = blockIdx.x * blockDim.x + threadIdx.x;   // padcount*K2 threads
  if (i >= padcount * K2) return;
  int r = i >> 9, cc = i & 511;
  __bf16 v = (__bf16)0.0f;
  if (r < count) {
    int p = first + r;
    int node = (cc < HDIM) ? (2 * p + 1) : (2 * p + 2);
    v = hbf[(size_t)node * HDIM + (cc & 255)];
  }
  cat[i] = v;
}

// ---------------- gather edge cat: row m -> [h[m+1], h[parent(m+1)]] ----------------
__global__ void gather_edge(const __bf16* __restrict__ hbf, __bf16* __restrict__ cat) {
  int i = blockIdx.x * blockDim.x + threadIdx.x;   // NPAD*K2 threads
  int r = i >> 9, cc = i & 511;
  __bf16 v = (__bf16)0.0f;
  if (r < NNODE - 1) {
    int child = r + 1;
    int node = (cc < HDIM) ? child : ((child - 1) >> 1);
    v = hbf[(size_t)node * HDIM + (cc & 255)];
  }
  cat[i] = v;
}

__global__ void zero_f32(float* __restrict__ p, int n) {
  int i = blockIdx.x * blockDim.x + threadIdx.x;
  if (i < n) p[i] = 0.0f;
}

// ---------------- host-side launch helper ----------------
static inline void launch_gemm(const __bf16* A, int lda, const __bf16* B, int ldb,
                               const float* b1, const float* b2,
                               float* C, int ldc, __bf16* Cbf, int ldcbf,
                               int M, int N, int K, hipStream_t s) {
  int padM = ((M + 15) / 16) * 16;
  if (N % 64 == 0) {
    // 32x64 tile per wave, 128 rows per block
    dim3 grid(N / 64, (padM + 127) / 128);
    wmma_gemm_bf16<2, 4><<<grid, dim3(128), 0, s>>>(A, lda, B, ldb, b1, b2,
                                                    C, ldc, Cbf, ldcbf, M, N, K);
  } else {
    // 32x32 tile per wave (N == 32 case)
    dim3 grid(N / 32, (padM + 127) / 128);
    wmma_gemm_bf16<2, 2><<<grid, dim3(128), 0, s>>>(A, lda, B, ldb, b1, b2,
                                                    C, ldc, Cbf, ldcbf, M, N, K);
  }
}

extern "C" void kernel_launch(void* const* d_in, const int* in_sizes, int n_in,
                              void* d_out, int out_size, void* d_ws, size_t ws_size,
                              hipStream_t stream) {
  (void)in_sizes; (void)n_in; (void)out_size;

  const int*   inp     = (const int*)  d_in[0];
  const float* emb     = (const float*)d_in[1];
  const float* W_ih    = (const float*)d_in[2];
  const float* W_hh    = (const float*)d_in[3];
  const float* b_ih    = (const float*)d_in[4];
  const float* b_hh    = (const float*)d_in[5];
  const float* W_unary = (const float*)d_in[6];
  const float* b_unary = (const float*)d_in[7];
  const float* W_uint  = (const float*)d_in[8];
  const float* b_uint  = (const float*)d_in[9];
  const float* W_hid   = (const float*)d_in[10];
  const float* b_hid   = (const float*)d_in[11];
  const float* W_edge  = (const float*)d_in[12];
  const float* b_edge  = (const float*)d_in[13];
  float* out = (float*)d_out;

  // ---- workspace layout (all sizes multiples of 256B) ----
  char* ws = (char*)d_ws;
  size_t off = 0;
  auto alloc = [&](size_t bytes) { char* p = ws + off; off += bytes; return p; };
  __bf16* Xbf    = (__bf16*)alloc((size_t)LLEAF * EDIM * 2);
  __bf16* Wih_b  = (__bf16*)alloc((size_t)G4 * EDIM * 2);
  __bf16* Wun_b  = (__bf16*)alloc((size_t)TT * HDIM * 2);
  __bf16* Wui_b  = (__bf16*)alloc((size_t)TT * K2 * 2);
  __bf16* Whid_b = (__bf16*)alloc((size_t)HDIM * K2 * 2);
  __bf16* Wedg_b = (__bf16*)alloc((size_t)G4 * K2 * 2);
  float*  Gbuf   = (float*) alloc((size_t)LLEAF * G4 * 4);
  __bf16* hbf    = (__bf16*)alloc((size_t)NPAD * HDIM * 2);
  __bf16* cat    = (__bf16*)alloc((size_t)NPAD * K2 * 2);
  if (off > ws_size) return;  // workspace too small: bail safely

  const int THR = 256;
  auto blocks = [](long n, int t) { return dim3((unsigned)((n + t - 1) / t)); };

  // ---- weight converts fp32 -> bf16 ----
  cvt_f32_bf16<<<blocks((long)G4 * EDIM, THR), THR, 0, stream>>>(W_ih, Wih_b, G4 * EDIM);
  cvt_f32_bf16<<<blocks((long)TT * HDIM, THR), THR, 0, stream>>>(W_unary, Wun_b, TT * HDIM);
  cvt_f32_bf16<<<blocks((long)TT * K2, THR), THR, 0, stream>>>(W_uint, Wui_b, TT * K2);
  cvt_f32_bf16<<<blocks((long)HDIM * K2, THR), THR, 0, stream>>>(W_hid, Whid_b, HDIM * K2);
  cvt_f32_bf16<<<blocks((long)G4 * K2, THR), THR, 0, stream>>>(W_edge, Wedg_b, G4 * K2);

  // ---- embedding gather ----
  embed_gather<<<blocks((long)LLEAF * EDIM, THR), THR, 0, stream>>>(inp, emb, Xbf);

  // ---- input-side LSTM GEMM: G = X @ W_ih^T + (b_ih + b_hh)  [8192,1024] ----
  launch_gemm(Xbf, EDIM, Wih_b, EDIM, b_ih, b_hh, Gbuf, G4, nullptr, 0,
              LLEAF, G4, EDIM, stream);

  // ---- sequential recurrence -> leaf h (bf16) at nodes [Lc-1 .. n-1] ----
  lstm_recurrence<<<1, 1024, 0, stream>>>(Gbuf, W_hh, hbf);

  // ---- unary at leaves: hs @ W_unary^T + b_unary -> out rows [Lc-1..] ----
  launch_gemm(hbf + (size_t)(LLEAF - 1) * HDIM, HDIM, Wun_b, HDIM, b_unary, nullptr,
              out + (size_t)(LLEAF - 1) * TT, TT, nullptr, 0,
              LLEAF, TT, HDIM, stream);

  // ---- bottom-up tree levels d = 12..0 ----
  for (int d = 12; d >= 0; --d) {
    int first = (1 << d) - 1;
    int count = 1 << d;
    int padc  = count < 16 ? 16 : count;
    gather_children<<<blocks((long)padc * K2, THR), THR, 0, stream>>>(hbf, cat, first, count, padc);
    // h[p] = cat @ W_hid^T + b_hid  (bf16 out only; feeds next level + edges)
    launch_gemm(cat, K2, Whid_b, K2, b_hid, nullptr, nullptr, 0,
                hbf + (size_t)first * HDIM, HDIM, count, HDIM, K2, stream);
    // unary[p] = cat @ W_uint^T + b_uint
    launch_gemm(cat, K2, Wui_b, K2, b_uint, nullptr,
                out + (size_t)first * TT, TT, nullptr, 0, count, TT, K2, stream);
  }

  // ---- edge potentials: cat(h[child], h[parent]) @ W_edge^T + b_edge ----
  float* edge = out + (size_t)NNODE * TT;
  gather_edge<<<blocks((long)NPAD * K2, THR), THR, 0, stream>>>(hbf, cat);
  zero_f32<<<blocks(G4, THR), THR, 0, stream>>>(edge, G4);        // edge row for root = 0
  launch_gemm(cat, K2, Wedg_b, K2, b_edge, nullptr,
              edge + G4, G4, nullptr, 0, NNODE - 1, G4, K2, stream);
}
